// GNN_Multihead_Attention_50242527429329
// MI455X (gfx1250) — matrix operand order repaired and checked by reading, not compiled, not run on hardware
//
#include <hip/hip_runtime.h>
#include <cstdint>
#include <cstddef>

// ---------------- types / constants ----------------
typedef __attribute__((ext_vector_type(16))) __bf16        v16bf;
typedef __attribute__((ext_vector_type(8)))  float         v8f;
typedef __attribute__((ext_vector_type(8)))  unsigned int  v8u;

#define D_IN      256
#define HC        256          // HEADS*HID
#define HID       64
#define HEADS     4
#define OUT_F     86
#define NEG_SLOPE 0.2f
#define EPS_BN    1e-5f

// ---------------- helpers ----------------
__device__ __forceinline__ unsigned bf16r(float f) {
  unsigned u = __float_as_uint(f);
  return (u + 0x7FFFu + ((u >> 16) & 1u)) >> 16;   // round-to-nearest-even
}
__device__ __forceinline__ unsigned bf16pk(float lo, float hi) {
  return bf16r(lo) | (bf16r(hi) << 16);
}
// float -> uint key with float ordering (for atomicMax-based segment max)
__device__ __forceinline__ unsigned fkey(float f) {
  unsigned u = __float_as_uint(f);
  return (u & 0x80000000u) ? ~u : (u | 0x80000000u);
}
__device__ __forceinline__ float fdec(unsigned k) {
  unsigned u = (k & 0x80000000u) ? (k ^ 0x80000000u) : ~k;
  return __uint_as_float(u);
}
__device__ __forceinline__ void edge_sd(const long long* __restrict__ ei,
                                        long long E, long long e, int& s, int& d) {
  if (e < E) { s = (int)ei[e]; d = (int)ei[E + e]; }
  else       { s = d = (int)(e - E); }              // appended self loops
}

// ---------------- fill kernels ----------------
__global__ void fill_f32(float* p, float v, size_t count) {
  size_t t = (size_t)blockIdx.x * blockDim.x + threadIdx.x;
  if (t < count) p[t] = v;
}
__global__ void fill_u32(unsigned* p, unsigned v, size_t count) {
  size_t t = (size_t)blockIdx.x * blockDim.x + threadIdx.x;
  if (t < count) p[t] = v;
}

// ---------------- f32 -> packed bf16 pairs ----------------
__global__ void f32_to_bf16(const float* __restrict__ X, unsigned* __restrict__ Y,
                            size_t npairs) {
  size_t t = (size_t)blockIdx.x * blockDim.x + threadIdx.x;
  if (t >= npairs) return;
  float2 f = ((const float2*)X)[t];
  Y[t] = bf16pk(f.x, f.y);
}

// ---------------- swizzle W[256,256] f32 into B-fragment order, packed bf16 -------------
// Ws layout: [tileN(16)][kt(8)][lane(32)][j(8)] dwords; dword j = bf16{W[kr+2j][col],
// W[kr+2j+1][col]} with col = tileN*16 + (lane&15), kr = kt*32 + (lane>=16 ? 16 : 0).
__global__ void wswizzle(const float* __restrict__ W, unsigned* __restrict__ Ws) {
  int t = blockIdx.x * blockDim.x + threadIdx.x;    // 16*8*32 = 4096 threads
  if (t >= 4096) return;
  int lane = t & 31, kt = (t >> 5) & 7, tn = t >> 8;
  int col = tn * 16 + (lane & 15);
  int kr  = kt * 32 + ((lane >> 4) * 16);
  unsigned o[8];
#pragma unroll
  for (int j = 0; j < 8; ++j)
    o[j] = bf16pk(W[(size_t)(kr + 2 * j) * HC + col],
                  W[(size_t)(kr + 2 * j + 1) * HC + col]);
  uint4* dst = (uint4*)(Ws + (size_t)t * 8);
  dst[0] = make_uint4(o[0], o[1], o[2], o[3]);
  dst[1] = make_uint4(o[4], o[5], o[6], o[7]);
}

// ---------------- bf16 WMMA GEMM: C[nrows,256] = A[nrows,256] @ W[256,256] --------------
// A is pre-converted packed bf16 (row-major, 128 dwords/row). Ws is pre-swizzled bf16.
// grid = (ceil(tilesM/8), 16), block = 256 (8 waves, one 16x16 tile each; all waves of a
// block share the same 16-column W strip, staged once in LDS).
__global__ __launch_bounds__(256) void gemm_bf16_256(const unsigned* __restrict__ Abf,
                                                     const unsigned* __restrict__ Ws,
                                                     float* __restrict__ C, int nrows) {
  __shared__ unsigned ldsW[2048];                   // 8KB: one tileN strip of Ws
  {
    const uint4* src = (const uint4*)(Ws + (size_t)blockIdx.y * 2048 + threadIdx.x * 8);
    uint4* dst = (uint4*)(ldsW + threadIdx.x * 8);
    dst[0] = src[0];
    dst[1] = src[1];
  }
  __syncthreads();

  const int wave   = threadIdx.x >> 5;
  const int lane   = threadIdx.x & 31;
  const int tilesM = (nrows + 15) >> 4;
  const int tileM  = blockIdx.x * 8 + wave;
  if (tileM >= tilesM) return;                      // wave-uniform: EXEC stays all-ones

  const int hiHalf = lane >> 4;                     // 0 | 1
  const int l15    = lane & 15;
  const int rowA   = tileM * 16 + l15;              // A-frag row for this lane
  const unsigned* arow = Abf + (size_t)rowA * 128 + hiHalf * 8;  // + kbA/2 dwords

  v8f acc = {0.f, 0.f, 0.f, 0.f, 0.f, 0.f, 0.f, 0.f};

#pragma unroll
  for (int kt = 0; kt < 8; ++kt) {                  // K = 8 * 32
    // A fragment: elems 0..15 are k = 16*hiHalf + 0..15 (one contiguous 32B run of bf16)
    uint4 a0 = *(const uint4*)(arow + kt * 16);
    uint4 a1 = *(const uint4*)(arow + kt * 16 + 4);
    // B fragment: 32B contiguous per lane in swizzled LDS strip
    const unsigned* bp = ldsW + ((kt * 32 + lane) * 8);
    uint4 b0 = *(const uint4*)(bp);
    uint4 b1 = *(const uint4*)(bp + 4);
    v8u au = {a0.x, a0.y, a0.z, a0.w, a1.x, a1.y, a1.z, a1.w};
    v8u bu = {b0.x, b0.y, b0.z, b0.w, b1.x, b1.y, b1.z, b1.w};
    v16bf a = __builtin_bit_cast(v16bf, au);
    v16bf b = __builtin_bit_cast(v16bf, bu);
    acc = __builtin_amdgcn_wmma_f32_16x16x32_bf16(false, a, false, b, (short)0, acc,
                                                  false, false);
  }

  // C/D layout: lane -> col l15, VGPR v -> row v + hiHalf*8
  const int col = blockIdx.y * 16 + l15;
  const int r0  = tileM * 16 + hiHalf * 8;
#pragma unroll
  for (int v = 0; v < 8; ++v) {
    int r = r0 + v;
    if (r < nrows) C[(size_t)r * HC + col] = acc[v];
  }
}

// ---------------- attention coefficients: a_s[n,h], a_d[n,h] ----------------
__global__ void attn_coef(const float* __restrict__ h, const float* __restrict__ att_s,
                          const float* __restrict__ att_d, float* __restrict__ a_s,
                          float* __restrict__ a_d, int n_nodes) {
  int tid = blockIdx.x * blockDim.x + threadIdx.x;
  if (tid >= n_nodes * HEADS) return;
  int n = tid >> 2, hd = tid & 3;
  const float* hp = h + (size_t)n * HC + hd * HID;
  const float* sp = att_s + hd * HID;
  const float* dp = att_d + hd * HID;
  float as = 0.f, ad = 0.f;
#pragma unroll
  for (int c = 0; c < HID; c += 4) {
    float4 hv = *(const float4*)(hp + c);
    float4 sv = *(const float4*)(sp + c);
    float4 dv = *(const float4*)(dp + c);
    as += hv.x * sv.x + hv.y * sv.y + hv.z * sv.z + hv.w * sv.w;
    ad += hv.x * dv.x + hv.y * dv.y + hv.z * dv.z + hv.w * dv.w;
  }
  a_s[tid] = as;
  a_d[tid] = ad;
}

// ---------------- edge pass 1: leaky-relu logits + segment max ----------------
__global__ void edge_logit_max(const long long* __restrict__ ei, long long E, long long EE,
                               const float* __restrict__ a_s, const float* __restrict__ a_d,
                               float* __restrict__ ev, unsigned* __restrict__ mk) {
  long long e = (long long)blockIdx.x * blockDim.x + threadIdx.x;
  if (e >= EE) return;
  int s, d; edge_sd(ei, E, e, s, d);
  float4 as = *(const float4*)(a_s + (size_t)s * 4);
  float4 ad = *(const float4*)(a_d + (size_t)d * 4);
  float l[4] = {as.x + ad.x, as.y + ad.y, as.z + ad.z, as.w + ad.w};
#pragma unroll
  for (int hd = 0; hd < 4; ++hd) {
    float v = l[hd];
    v = v >= 0.f ? v : NEG_SLOPE * v;
    l[hd] = v;
    atomicMax(mk + (size_t)d * 4 + hd, fkey(v));
  }
  *(float4*)(ev + (size_t)e * 4) = make_float4(l[0], l[1], l[2], l[3]);
}

// ---------------- edge pass 2: exp(logit - max) + segment sum ----------------
__global__ void edge_exp_sum(const long long* __restrict__ ei, long long E, long long EE,
                             float* __restrict__ ev, const unsigned* __restrict__ mk,
                             float* __restrict__ ssum) {
  long long e = (long long)blockIdx.x * blockDim.x + threadIdx.x;
  if (e >= EE) return;
  int s, d; edge_sd(ei, E, e, s, d);
  float4 l = *(const float4*)(ev + (size_t)e * 4);
  float vals[4] = {l.x, l.y, l.z, l.w};
#pragma unroll
  for (int hd = 0; hd < 4; ++hd) {
    float m  = fdec(mk[(size_t)d * 4 + hd]);
    float ex = __expf(vals[hd] - m);
    vals[hd] = ex;
    atomicAdd(ssum + (size_t)d * 4 + hd, ex);
  }
  *(float4*)(ev + (size_t)e * 4) = make_float4(vals[0], vals[1], vals[2], vals[3]);
}

// ---------------- edge pass 3: weighted aggregate (one block per edge) ----------------
__global__ __launch_bounds__(256) void edge_aggregate(const long long* __restrict__ ei,
                                                      long long E,
                                                      const float* __restrict__ h,
                                                      const float* __restrict__ ev,
                                                      const float* __restrict__ ssum,
                                                      float* __restrict__ agg) {
  long long e = blockIdx.x;
  int c = threadIdx.x;
  int s, d; edge_sd(ei, E, e, s, d);
  int hd = c >> 6;
  float alpha = ev[(size_t)e * 4 + hd] / (ssum[(size_t)d * 4 + hd] + 1e-16f);
  atomicAdd(agg + (size_t)d * HC + c, h[(size_t)s * HC + c] * alpha);
}

// ---------------- batchnorm column stats (blockDim == C) ----------------
__global__ void bn_stats(const float* __restrict__ X, int nrows, int C,
                         float* __restrict__ sum, float* __restrict__ sumsq) {
  int c = threadIdx.x;
  int base = blockIdx.x * 64;
  float s1 = 0.f, s2 = 0.f;
  for (int r = 0; r < 64; ++r) {
    int row = base + r;
    if (row < nrows) {
      float v = X[(size_t)row * C + c];
      s1 += v;
      s2 += v * v;
    }
  }
  atomicAdd(sum + c, s1);
  atomicAdd(sumsq + c, s2);
}

__global__ void bn_finalize(const float* __restrict__ sum, const float* __restrict__ sumsq,
                            const float* __restrict__ g, const float* __restrict__ be,
                            int nrows, float* __restrict__ scale, float* __restrict__ shift) {
  int c = threadIdx.x;
  float inv  = 1.f / (float)nrows;
  float mean = sum[c] * inv;
  float var  = sumsq[c] * inv - mean * mean;
  float sc   = g[c] * rsqrtf(var + EPS_BN);
  scale[c] = sc;
  shift[c] = be[c] - mean * sc;               // bias b cancels inside batchnorm
}

// ---------------- BN + PReLU (layer 1, C=256) -> packed bf16 output ----------------
// Output feeds only GEMM2, so write it directly in the bf16 A format.
__global__ void bn_prelu_bf16(const float* __restrict__ X, const float* __restrict__ scale,
                              const float* __restrict__ shift, const float* __restrict__ pw,
                              unsigned* __restrict__ Ybf, size_t npairs) {
  size_t t = (size_t)blockIdx.x * blockDim.x + threadIdx.x;
  if (t >= npairs) return;
  size_t i = t * 2;
  int c0 = (int)(i & (HC - 1));                     // even channel
  float p  = pw[0];
  float2 v = ((const float2*)X)[t];
  float y0 = scale[c0] * v.x + shift[c0];
  float y1 = scale[c0 + 1] * v.y + shift[c0 + 1];
  y0 = y0 >= 0.f ? y0 : p * y0;
  y1 = y1 >= 0.f ? y1 : p * y1;
  Ybf[t] = bf16pk(y0, y1);
}

// ---------------- head mean (concat=False) ----------------
__global__ void head_mean(const float* __restrict__ agg, float* __restrict__ hm,
                          int n_nodes) {
  int t = blockIdx.x * blockDim.x + threadIdx.x;
  if (t >= n_nodes * HID) return;
  int n = t >> 6, c = t & 63;
  const float* p = agg + (size_t)n * HC;
  hm[t] = 0.25f * (p[c] + p[64 + c] + p[128 + c] + p[192 + c]);
}

// ---------------- BN + ELU (layer 2, C=64, in-place) ----------------
__global__ void bn_elu(float* __restrict__ X, const float* __restrict__ scale,
                       const float* __restrict__ shift, size_t count) {
  size_t t = (size_t)blockIdx.x * blockDim.x + threadIdx.x;
  if (t >= count) return;
  int c = (int)(t & (HID - 1));
  float y = scale[c] * X[t] + shift[c];
  X[t] = y > 0.f ? y : expm1f(y);
}

// ---------------- final fc: out[n,86] = z[n,64] @ fcW + fcb ----------------
__global__ void fc_kernel(const float* __restrict__ Z, const float* __restrict__ Wf,
                          const float* __restrict__ bf, float* __restrict__ out,
                          int n_nodes) {
  long long t = (long long)blockIdx.x * blockDim.x + threadIdx.x;
  if (t >= (long long)n_nodes * OUT_F) return;
  int n = (int)(t / OUT_F), o = (int)(t % OUT_F);
  const float* z = Z + (size_t)n * HID;
  float acc = bf[o];
#pragma unroll 8
  for (int c = 0; c < HID; ++c) acc += z[c] * Wf[c * OUT_F + o];
  out[t] = acc;
}

// ---------------- launch ----------------
extern "C" void kernel_launch(void* const* d_in, const int* in_sizes, int n_in,
                              void* d_out, int out_size, void* d_ws, size_t ws_size,
                              hipStream_t stream) {
  (void)n_in; (void)out_size; (void)ws_size;
  const float*     x   = (const float*)d_in[0];
  const long long* ei  = (const long long*)d_in[1];
  const float*     W1  = (const float*)d_in[2];
  const float*     as1 = (const float*)d_in[3];
  const float*     ad1 = (const float*)d_in[4];
  /* d_in[5] = b1 : cancelled by batchnorm */
  const float*     W2  = (const float*)d_in[6];
  const float*     as2 = (const float*)d_in[7];
  const float*     ad2 = (const float*)d_in[8];
  /* d_in[9] = b2 : cancelled by batchnorm */
  const float*     g1  = (const float*)d_in[10];
  const float*     be1 = (const float*)d_in[11];
  const float*     g2  = (const float*)d_in[12];
  const float*     be2 = (const float*)d_in[13];
  const float*     pw  = (const float*)d_in[14];
  const float*     fcW = (const float*)d_in[15];
  const float*     fcb = (const float*)d_in[16];
  float*           out = (float*)d_out;

  const int       Nn = in_sizes[0] / D_IN;
  const long long E  = in_sizes[1] / 2;
  const long long EE = E + Nn;

  // workspace carve-up (256B aligned)
  char* ws = (char*)d_ws;
  size_t off = 0;
  auto alloc = [&](size_t bytes) -> char* {
    char* p = ws + off;
    off += (bytes + 255) & ~(size_t)255;
    return p;
  };
  float*    bufA = (float*)alloc((size_t)Nn * HC * 4);    // h1 -> agg2
  float*    bufB = (float*)alloc((size_t)Nn * HC * 4);    // agg1 -> h2
  unsigned* abf  = (unsigned*)alloc((size_t)Nn * 128 * 4);// x_bf -> hact_bf (packed bf16)
  float*    ev   = (float*)alloc((size_t)EE * 4 * 4);     // logits/exp -> hm/z
  unsigned* w1s  = (unsigned*)alloc(4096 * 8 * 4);        // swizzled bf16 W1
  unsigned* w2s  = (unsigned*)alloc(4096 * 8 * 4);        // swizzled bf16 W2
  float*    a_s  = (float*)alloc((size_t)Nn * 4 * 4);
  float*    a_d  = (float*)alloc((size_t)Nn * 4 * 4);
  unsigned* mk   = (unsigned*)alloc((size_t)Nn * 4 * 4);
  float*    ssum = (float*)alloc((size_t)Nn * 4 * 4);
  float*    csum = (float*)alloc(HC * 4);
  float*    csq  = (float*)alloc(HC * 4);
  float*    bsc  = (float*)alloc(HC * 4);
  float*    bsh  = (float*)alloc(HC * 4);

  const unsigned NEG_KEY = 0x007FFFFFu;  // fkey(-inf)
  auto cdiv = [](long long a, long long b) -> unsigned { return (unsigned)((a + b - 1) / b); };

  const int tilesM = (Nn + 15) / 16;
  dim3 ggrid((tilesM + 7) / 8, HC / 16);

  // ================= Layer 1 =================
  f32_to_bf16<<<cdiv((long long)Nn * 128, 256), 256, 0, stream>>>(x, abf, (size_t)Nn * 128);
  wswizzle<<<16, 256, 0, stream>>>(W1, w1s);
  wswizzle<<<16, 256, 0, stream>>>(W2, w2s);
  gemm_bf16_256<<<ggrid, 256, 0, stream>>>(abf, w1s, bufA, Nn);                   // h1
  attn_coef<<<cdiv((long long)Nn * 4, 256), 256, 0, stream>>>(bufA, as1, ad1, a_s, a_d, Nn);
  fill_u32<<<cdiv((long long)Nn * 4, 256), 256, 0, stream>>>(mk, NEG_KEY, (size_t)Nn * 4);
  fill_f32<<<cdiv((long long)Nn * 4, 256), 256, 0, stream>>>(ssum, 0.f, (size_t)Nn * 4);
  fill_f32<<<cdiv((long long)Nn * HC, 256), 256, 0, stream>>>(bufB, 0.f, (size_t)Nn * HC);
  edge_logit_max<<<cdiv(EE, 256), 256, 0, stream>>>(ei, E, EE, a_s, a_d, ev, mk);
  edge_exp_sum<<<cdiv(EE, 256), 256, 0, stream>>>(ei, E, EE, ev, mk, ssum);
  edge_aggregate<<<dim3((unsigned)EE), 256, 0, stream>>>(ei, E, bufA, ev, ssum, bufB);
  fill_f32<<<1, 256, 0, stream>>>(csum, 0.f, HC);
  fill_f32<<<1, 256, 0, stream>>>(csq, 0.f, HC);
  bn_stats<<<cdiv(Nn, 64), HC, 0, stream>>>(bufB, Nn, HC, csum, csq);
  bn_finalize<<<1, HC, 0, stream>>>(csum, csq, g1, be1, Nn, bsc, bsh);
  bn_prelu_bf16<<<cdiv((long long)Nn * 128, 256), 256, 0, stream>>>(bufB, bsc, bsh, pw, abf,
                                                                    (size_t)Nn * 128);

  // ================= Layer 2 =================
  gemm_bf16_256<<<ggrid, 256, 0, stream>>>(abf, w2s, bufB, Nn);                   // h2
  attn_coef<<<cdiv((long long)Nn * 4, 256), 256, 0, stream>>>(bufB, as2, ad2, a_s, a_d, Nn);
  fill_u32<<<cdiv((long long)Nn * 4, 256), 256, 0, stream>>>(mk, NEG_KEY, (size_t)Nn * 4);
  fill_f32<<<cdiv((long long)Nn * 4, 256), 256, 0, stream>>>(ssum, 0.f, (size_t)Nn * 4);
  fill_f32<<<cdiv((long long)Nn * HC, 256), 256, 0, stream>>>(bufA, 0.f, (size_t)Nn * HC);
  edge_logit_max<<<cdiv(EE, 256), 256, 0, stream>>>(ei, E, EE, a_s, a_d, ev, mk);
  edge_exp_sum<<<cdiv(EE, 256), 256, 0, stream>>>(ei, E, EE, ev, mk, ssum);
  edge_aggregate<<<dim3((unsigned)EE), 256, 0, stream>>>(ei, E, bufB, ev, ssum, bufA);
  head_mean<<<cdiv((long long)Nn * HID, 256), 256, 0, stream>>>(bufA, ev, Nn);    // hm in ev
  fill_f32<<<1, 256, 0, stream>>>(csum, 0.f, HC);
  fill_f32<<<1, 256, 0, stream>>>(csq, 0.f, HC);
  bn_stats<<<cdiv(Nn, 64), HID, 0, stream>>>(ev, Nn, HID, csum, csq);
  bn_finalize<<<1, HID, 0, stream>>>(csum, csq, g2, be2, Nn, bsc, bsh);
  bn_elu<<<cdiv((long long)Nn * HID, 256), 256, 0, stream>>>(ev, bsc, bsh, (size_t)Nn * HID);
  fc_kernel<<<cdiv((long long)Nn * OUT_F, 256), 256, 0, stream>>>(ev, fcW, fcb, out, Nn);
}
